// NGCF_18657337934509
// MI455X (gfx1250) — matrix operand order repaired and checked by reading, not compiled, not run on hardware
//
#include <hip/hip_runtime.h>
#include <hip/hip_bf16.h>
#include <math.h>

#define NN   4096
#define EE   262144
#define ESL  (EE + NN)
#define NCLS 41476

// ---------------- bf16 helpers ----------------
typedef __attribute__((ext_vector_type(16))) __bf16 v16bf;
typedef __attribute__((ext_vector_type(8)))  float  v8f;

union FragAB { uint4 q[2]; v16bf v; };

__device__ __forceinline__ unsigned short f2bf(float f) {
  unsigned int u = __float_as_uint(f);
  u += 0x7FFFu + ((u >> 16) & 1u);          // round-to-nearest-even
  return (unsigned short)(u >> 16);
}

// elementwise f32 -> bf16, 8 elements/thread
__global__ void k_f2bf(const float* __restrict__ x, unsigned short* __restrict__ y, int n8) {
  int t = blockIdx.x * 256 + threadIdx.x;
  if (t >= n8) return;
  float4 a = *(const float4*)&x[(size_t)t * 8];
  float4 b = *(const float4*)&x[(size_t)t * 8 + 4];
  union { unsigned short h[8]; uint4 q; } u;
  u.h[0] = f2bf(a.x); u.h[1] = f2bf(a.y); u.h[2] = f2bf(a.z); u.h[3] = f2bf(a.w);
  u.h[4] = f2bf(b.x); u.h[5] = f2bf(b.y); u.h[6] = f2bf(b.z); u.h[7] = f2bf(b.w);
  *(uint4*)&y[(size_t)t * 8] = u.q;
}

// W[K][N] f32  ->  WT[N][K] bf16  (tiled 32K x 64N transpose through LDS)
__global__ __launch_bounds__(256)
void k_wt(const float* __restrict__ W, unsigned short* __restrict__ WT, int K, int N) {
  __shared__ __align__(16) unsigned short T[64 * 40];   // [n][k], padded stride 40
  const int tid = threadIdx.x;
  const int k0 = blockIdx.y * 32, n0 = blockIdx.x * 64;
#pragma unroll
  for (int it = 0; it < 2; ++it) {
    int i  = tid + it * 256;            // 0..511 float4 chunks (32 rows x 16 chunks)
    int kk = i >> 4;
    int nb = (i & 15) << 2;
    int n  = n0 + nb;
    float4 v = make_float4(0.f, 0.f, 0.f, 0.f);
    if (n + 3 < N) v = *(const float4*)&W[(size_t)(k0 + kk) * N + n];
    T[(nb + 0) * 40 + kk] = f2bf(v.x);
    T[(nb + 1) * 40 + kk] = f2bf(v.y);
    T[(nb + 2) * 40 + kk] = f2bf(v.z);
    T[(nb + 3) * 40 + kk] = f2bf(v.w);
  }
  __syncthreads();
  int r = tid >> 2, cb = (tid & 3) << 3;   // 64 rows x 4 chunks of 8 halves
  int n = n0 + r;
  if (n < N) {
    uint4 v = *(const uint4*)&T[r * 40 + cb];
    *(uint4*)&WT[(size_t)n * K + k0 + cb] = v;
  }
}

// ---------------- WMMA bf16 GEMM ----------------
#define BM 128
#define BN 128
#define BK 32
#define LSTR 40   // LDS row stride in halves (32 + 8 pad)

// C[M,N] = act( A[M,K](bf16) * BT[N,K](bf16)^T + bias + (beta? C : 0) )
// Requires M%128==0, K%32==0, N%4==0.
__global__ __launch_bounds__(256)
void k_gemm_bf16(const unsigned short* __restrict__ A, const unsigned short* __restrict__ BT,
                 const float* __restrict__ bias, float* __restrict__ C,
                 int M, int N, int K, int beta, int act)
{
  __shared__ __align__(16) unsigned short As[BM * LSTR];
  __shared__ __align__(16) unsigned short Bs[BN * LSTR];

  const int tid  = threadIdx.x;
  const int lane = tid & 31;
  const int wave = tid >> 5;
  const int wm   = wave & 3;     // 4 waves down M  (32 rows each)
  const int wn   = wave >> 2;    // 2 waves across N (64 cols each)
  const int blockM = blockIdx.y * BM;
  const int blockN = blockIdx.x * BN;

  v8f acc[2][4];
#pragma unroll
  for (int i = 0; i < 2; ++i)
#pragma unroll
    for (int j = 0; j < 4; ++j)
#pragma unroll
      for (int r = 0; r < 8; ++r) acc[i][j][r] = 0.0f;

  for (int k0 = 0; k0 < K; k0 += BK) {
    // stage A tile (128 rows x 32 halves) : pure b128 copies
#pragma unroll
    for (int it = 0; it < 2; ++it) {
      int i  = tid + it * 256;       // 0..511 chunks of 8 halves
      int r  = i >> 2;
      int cb = (i & 3) << 3;
      uint4 v = *(const uint4*)&A[(size_t)(blockM + r) * K + k0 + cb];
      *(uint4*)&As[r * LSTR + cb] = v;
    }
    // stage B tile (128 N-rows x 32 halves) from pre-transposed weights
#pragma unroll
    for (int it = 0; it < 2; ++it) {
      int i  = tid + it * 256;
      int r  = i >> 2;
      int cb = (i & 3) << 3;
      int n  = blockN + r;
      uint4 v = make_uint4(0u, 0u, 0u, 0u);
      if (n < N) v = *(const uint4*)&BT[(size_t)n * K + k0 + cb];
      *(uint4*)&Bs[r * LSTR + cb] = v;
    }
    __syncthreads();

    // A fragment: lanes 0-15 hold K kb..kb+7 / kb+16..kb+23 (kb = 0|8)
    FragAB af[2], bfr[4];
    const int arow = lane & 15;
    const int akb  = (lane >> 4) << 3;
#pragma unroll
    for (int mi = 0; mi < 2; ++mi) {
      const unsigned short* p = &As[(wm * 32 + mi * 16 + arow) * LSTR + akb];
      af[mi].q[0] = *(const uint4*)(p);
      af[mi].q[1] = *(const uint4*)(p + 16);
    }
    // B fragment: lanes hold K kb..kb+7 / kb+16..kb+23 (kb = 0|8), col = lane&15
    const int bcol = lane & 15;
    const int bkb  = (lane >> 4) << 3;
#pragma unroll
    for (int ni = 0; ni < 4; ++ni) {
      const unsigned short* p = &Bs[(wn * 64 + ni * 16 + bcol) * LSTR + bkb];
      bfr[ni].q[0] = *(const uint4*)(p);
      bfr[ni].q[1] = *(const uint4*)(p + 16);
    }

#pragma unroll
    for (int mi = 0; mi < 2; ++mi)
#pragma unroll
      for (int ni = 0; ni < 4; ++ni)
        acc[mi][ni] = __builtin_amdgcn_wmma_f32_16x16x32_bf16(
            false, af[mi].v, false, bfr[ni].v, (short)0, acc[mi][ni], false, false);

    __syncthreads();
  }

  // epilogue: lanes 0-15 -> M=r, lanes 16-31 -> M=r+8; N = lane&15
  const int rbase = (lane < 16) ? 0 : 8;
#pragma unroll
  for (int mi = 0; mi < 2; ++mi)
#pragma unroll
    for (int ni = 0; ni < 4; ++ni) {
      int col = blockN + wn * 64 + ni * 16 + (lane & 15);
      if (col >= N) continue;
      int row0 = blockM + wm * 32 + mi * 16 + rbase;
      float bv = bias ? bias[col] : 0.f;
#pragma unroll
      for (int r = 0; r < 8; ++r) {
        size_t idx = (size_t)(row0 + r) * N + col;
        float v = acc[mi][ni][r] + bv;
        if (beta) v += C[idx];
        if (act == 1) v = fmaxf(v, 0.f);
        C[idx] = v;
      }
    }
}

// ---------------- elementwise / graph kernels ----------------
__global__ void k_fill(float* p, float v, int n) {
  int t = blockIdx.x * 256 + threadIdx.x;
  if (t < n) p[t] = v;
}

__global__ void k_gather(const int* __restrict__ ui, const int* __restrict__ ii,
                         const float* __restrict__ ue, const float* __restrict__ ie,
                         float* __restrict__ out) {
  int t = blockIdx.x * 256 + threadIdx.x;           // NN*128 threads
  int n = t >> 7, j = t & 127;
  if (j < 64) {
    float4 v = *(const float4*)&ue[(size_t)ui[n] * 256 + j * 4];
    *(float4*)&out[(size_t)n * 512 + j * 4] = v;
  } else {
    int jj = j - 64;
    float4 v = *(const float4*)&ie[(size_t)ii[n] * 256 + jj * 4];
    *(float4*)&out[(size_t)n * 512 + 256 + jj * 4] = v;
  }
}

__global__ void k_build_sl(const int* __restrict__ row, const int* __restrict__ col,
                           int* __restrict__ rsl, int* __restrict__ csl) {
  int t = blockIdx.x * 256 + threadIdx.x;
  if (t < EE)      { rsl[t] = row[t]; csl[t] = col[t]; }
  else if (t < ESL){ rsl[t] = t - EE; csl[t] = t - EE; }
}

__global__ void k_deg(const int* __restrict__ idx, int nE, float* __restrict__ deg) {
  int e = blockIdx.x * 256 + threadIdx.x;
  if (e < nE) atomicAdd(&deg[idx[e]], 1.0f);
}

__global__ void k_dinv(const float* __restrict__ deg, float* __restrict__ dinv, int n) {
  int t = blockIdx.x * 256 + threadIdx.x;
  if (t < n) { float d = deg[t]; dinv[t] = d > 0.f ? rsqrtf(d) : 0.f; }
}

// out[col] += w * x[row];  w = wedge[e] (if wedge) else sgn*dinv[r]*dinv[c] (if dinv) else 1
__global__ void k_edge_agg(float* __restrict__ out, const float* __restrict__ x,
                           const int* __restrict__ row, const int* __restrict__ col,
                           const float* __restrict__ wedge, const float* __restrict__ dinv,
                           float sgn, int K4, int K, int nE) {
  int t = blockIdx.x * 256 + threadIdx.x;
  if (t >= nE * K4) return;
  int e = t / K4;
  int d = (t - e * K4) * 4;
  int r = row[e], c = col[e];
  float w = 1.f;
  if (dinv)  w = sgn * dinv[r] * dinv[c];
  if (wedge) w = wedge[e];
  float4 xv = *(const float4*)&x[(size_t)r * K + d];
  float* o = &out[(size_t)c * K + d];
  atomicAdd(o + 0, xv.x * w);
  atomicAdd(o + 1, xv.y * w);
  atomicAdd(o + 2, xv.z * w);
  atomicAdd(o + 3, xv.w * w);
}

__global__ void k_bn_stats(const float* __restrict__ x, float* __restrict__ mu,
                           float* __restrict__ rs, int K) {
  __shared__ float ss[256], sq[256];
  int c = blockIdx.x, tid = threadIdx.x;
  float s = 0.f, q = 0.f;
  for (int r = tid; r < NN; r += 256) {
    float v = x[(size_t)r * K + c];
    s += v; q += v * v;
  }
  ss[tid] = s; sq[tid] = q; __syncthreads();
  for (int o = 128; o; o >>= 1) {
    if (tid < o) { ss[tid] += ss[tid + o]; sq[tid] += sq[tid + o]; }
    __syncthreads();
  }
  if (!tid) {
    float m = ss[0] * (1.f / NN);
    float var = sq[0] * (1.f / NN) - m * m;     // biased var, matches x.var(0)
    mu[c] = m; rs[c] = rsqrtf(var + 1e-5f);
  }
}

__global__ void k_bn_apply_relu(const float* __restrict__ x, const float* __restrict__ mu,
                                const float* __restrict__ rs, const float* __restrict__ g,
                                const float* __restrict__ b, float* __restrict__ y, int K) {
  int t = blockIdx.x * 256 + threadIdx.x;     // NN*K threads
  int c = t % K;
  float v = (x[t] - mu[c]) * rs[c] * g[c] + b[c];
  y[t] = fmaxf(v, 0.f);
}

__global__ void k_sage_mean(float* __restrict__ msum, const float* __restrict__ cnt, int K) {
  int t = blockIdx.x * 256 + threadIdx.x;     // NN*K threads
  int n = t / K;
  msum[t] /= fmaxf(cnt[n], 1.f);
}

__global__ void k_bias_elu(float* __restrict__ x, const float* __restrict__ b, int K) {
  int t = blockIdx.x * 256 + threadIdx.x;     // NN*K threads
  int c = t % K;
  float v = x[t] + b[c];
  x[t] = v > 0.f ? v : expm1f(v);
}

__global__ void k_gat_logits(const float* __restrict__ xw, const float* __restrict__ asr,
                             const float* __restrict__ adt, float* __restrict__ als,
                             float* __restrict__ ald, int D) {
  int row  = blockIdx.x * 8 + (threadIdx.x >> 5);
  int lane = threadIdx.x & 31;
  float s = 0.f, d = 0.f;
  for (int j = lane; j < D; j += 32) {
    float v = xw[(size_t)row * D + j];
    s += v * asr[j]; d += v * adt[j];
  }
  for (int o = 16; o; o >>= 1) { s += __shfl_down(s, o, 32); d += __shfl_down(d, o, 32); }
  if (!lane) { als[row] = s; ald[row] = d; }
}

__device__ __forceinline__ void atomicMaxF(float* a, float v) {
  if (v >= 0.f) atomicMax((int*)a, __float_as_int(v));
  else          atomicMin((unsigned int*)a, __float_as_uint(v));
}

__global__ void k_gat_e(const float* __restrict__ als, const float* __restrict__ ald,
                        const int* __restrict__ row, const int* __restrict__ col,
                        float* __restrict__ ev, float* __restrict__ emax, int nE) {
  int e = blockIdx.x * 256 + threadIdx.x;
  if (e >= nE) return;
  float v = als[row[e]] + ald[col[e]];
  v = v > 0.f ? v : 0.2f * v;                  // leaky_relu(0.2)
  ev[e] = v;
  atomicMaxF(&emax[col[e]], v);
}

__global__ void k_gat_exp(float* __restrict__ ev, const int* __restrict__ col,
                          const float* __restrict__ emax, float* __restrict__ den, int nE) {
  int e = blockIdx.x * 256 + threadIdx.x;
  if (e >= nE) return;
  float v = expf(ev[e] - emax[col[e]]);
  ev[e] = v;
  atomicAdd(&den[col[e]], v);
}

__global__ void k_gat_alpha(float* __restrict__ ev, const int* __restrict__ col,
                            const float* __restrict__ den, int nE) {
  int e = blockIdx.x * 256 + threadIdx.x;
  if (e >= nE) return;
  ev[e] /= den[col[e]];
}

// ---------------- orchestration ----------------
static inline int cdiv(int a, int b) { return (a + b - 1) / b; }

extern "C" void kernel_launch(void* const* d_in, const int* in_sizes, int n_in,
                              void* d_out, int out_size, void* d_ws, size_t ws_size,
                              hipStream_t stream) {
  (void)in_sizes; (void)n_in; (void)out_size; (void)ws_size;

  const int*   user_idx = (const int*)d_in[0];
  const int*   item_idx = (const int*)d_in[1];
  const int*   edge     = (const int*)d_in[2];       // [2,E]: row = edge, col = edge+E
  const float* user_emb = (const float*)d_in[3];
  const float* item_emb = (const float*)d_in[4];
  const float* gcn0_w   = (const float*)d_in[5];
  // gcn0_b (d_in[6]) cancels exactly through training-mode BN
  const float* bn0_g    = (const float*)d_in[7];
  const float* bn0_b    = (const float*)d_in[8];
  const float* gcn1_w   = (const float*)d_in[9];
  // gcn1_b (d_in[10]) cancels through BN
  const float* bn1_g    = (const float*)d_in[11];
  const float* bn1_b    = (const float*)d_in[12];
  const float* sage_wl  = (const float*)d_in[13];
  const float* sage_bl  = (const float*)d_in[14];
  const float* sage_wr  = (const float*)d_in[15];
  const float* cheb_w0  = (const float*)d_in[16];
  const float* cheb_w1  = (const float*)d_in[17];
  const float* cheb_b   = (const float*)d_in[18];
  const float* gat0_w   = (const float*)d_in[19];
  const float* gat0_as  = (const float*)d_in[20];
  const float* gat0_ad  = (const float*)d_in[21];
  const float* gat0_b   = (const float*)d_in[22];
  const float* gat1_w   = (const float*)d_in[23];
  const float* gat1_as  = (const float*)d_in[24];
  const float* gat1_ad  = (const float*)d_in[25];
  const float* gat1_b   = (const float*)d_in[26];
  const float* pred_w   = (const float*)d_in[27];
  const float* pred_b   = (const float*)d_in[28];

  // workspace carve-up
  char* w = (char*)d_ws;
  const size_t BUF = (size_t)NN * 512;
  float* buf0 = (float*)w; w += BUF * 4;
  float* buf1 = (float*)w; w += BUF * 4;
  float* buf2 = (float*)w; w += BUF * 4;
  float* buf3 = (float*)w; w += BUF * 4;
  float* deg     = (float*)w; w += NN * 4;
  float* dinv_sl = (float*)w; w += NN * 4;
  float* dinv_r  = (float*)w; w += NN * 4;
  float* cnt     = (float*)w; w += NN * 4;
  float* als     = (float*)w; w += NN * 4;
  float* ald     = (float*)w; w += NN * 4;
  float* emax    = (float*)w; w += NN * 4;
  float* den     = (float*)w; w += NN * 4;
  float* mu      = (float*)w; w += 512 * 4;
  float* rs      = (float*)w; w += 512 * 4;
  int*   row_sl  = (int*)w;   w += (size_t)ESL * 4;
  int*   col_sl  = (int*)w;   w += (size_t)ESL * 4;
  float* ev      = (float*)w; w += (size_t)ESL * 4;
  // bf16 staging: activations + pre-transposed weights
  unsigned short* abuf    = (unsigned short*)w; w += BUF * 2;                 // [M][K] bf16
  unsigned short* wt_gcn0 = (unsigned short*)w; w += (size_t)512 * 512 * 2;   // [N=512][K=512]
  unsigned short* wt_gcn1 = (unsigned short*)w; w += (size_t)256 * 512 * 2;   // [256][512]
  unsigned short* wt_sl   = (unsigned short*)w; w += (size_t)256 * 256 * 2;
  unsigned short* wt_sr   = (unsigned short*)w; w += (size_t)256 * 256 * 2;
  unsigned short* wt_c0   = (unsigned short*)w; w += (size_t)256 * 256 * 2;
  unsigned short* wt_c1   = (unsigned short*)w; w += (size_t)256 * 256 * 2;
  unsigned short* wt_g0   = (unsigned short*)w; w += (size_t)256 * 256 * 2;
  unsigned short* wt_g1   = (unsigned short*)w; w += (size_t)256 * 256 * 2;
  unsigned short* wt_pred = (unsigned short*)w; w += (size_t)NCLS * 256 * 2;  // [41476][256]

  const dim3 B256(256);
  auto GEMM = [&](const unsigned short* A, const unsigned short* BT, const float* bias,
                  float* C, int M, int N, int K, int beta, int act) {
    dim3 g(cdiv(N, BN), M / BM);
    k_gemm_bf16<<<g, B256, 0, stream>>>(A, BT, bias, C, M, N, K, beta, act);
  };
  auto CVT = [&](const float* src, int nElem) {   // activation -> abuf (bf16)
    int n8 = nElem / 8;
    k_f2bf<<<cdiv(n8, 256), B256, 0, stream>>>(src, abuf, n8);
  };
  auto WTC = [&](const float* Wsrc, unsigned short* dst, int K, int N) {
    dim3 g(cdiv(N, 64), K / 32);
    k_wt<<<g, B256, 0, stream>>>(Wsrc, dst, K, N);
  };
  auto FILL = [&](float* p, float v, int n) {
    k_fill<<<cdiv(n, 256), B256, 0, stream>>>(p, v, n);
  };
  auto AGG = [&](float* out, const float* x, const int* r, const int* c,
                 const float* we, const float* dv, float sgn, int K, int nE) {
    int K4 = K / 4;
    k_edge_agg<<<cdiv(nE * K4, 256), B256, 0, stream>>>(out, x, r, c, we, dv, sgn, K4, K, nE);
  };

  // one-time: convert+transpose all weights to bf16 [N][K]
  WTC(gcn0_w, wt_gcn0, 512, 512);
  WTC(gcn1_w, wt_gcn1, 512, 256);
  WTC(sage_wl, wt_sl, 256, 256);
  WTC(sage_wr, wt_sr, 256, 256);
  WTC(cheb_w0, wt_c0, 256, 256);
  WTC(cheb_w1, wt_c1, 256, 256);
  WTC(gat0_w,  wt_g0, 256, 256);
  WTC(gat1_w,  wt_g1, 256, 256);
  WTC(pred_w,  wt_pred, 256, NCLS);

  // x0 = concat(user_emb[user_idx], item_emb[item_idx]) -> [4096,512]
  k_gather<<<NN * 128 / 256, B256, 0, stream>>>(user_idx, item_idx, user_emb, item_emb, buf0);

  // self-loop edge list (shared by GCN0/1 and both GATs)
  k_build_sl<<<cdiv(ESL, 256), B256, 0, stream>>>(edge, edge + EE, row_sl, col_sl);

  // GCN symmetric norm degrees (in-degree at col, +1 self loop)
  FILL(deg, 0.f, NN);
  k_deg<<<cdiv(ESL, 256), B256, 0, stream>>>(col_sl, ESL, deg);
  k_dinv<<<cdiv(NN, 256), B256, 0, stream>>>(deg, dinv_sl, NN);

  // ---- GCN0 (512 -> 512) + BN + ReLU
  CVT(buf0, NN * 512);
  GEMM(abuf, wt_gcn0, nullptr, buf1, NN, 512, 512, 0, 0);
  FILL(buf2, 0.f, NN * 512);
  AGG(buf2, buf1, row_sl, col_sl, nullptr, dinv_sl, 1.f, 512, ESL);
  k_bn_stats<<<512, B256, 0, stream>>>(buf2, mu, rs, 512);
  k_bn_apply_relu<<<NN * 512 / 256, B256, 0, stream>>>(buf2, mu, rs, bn0_g, bn0_b, buf0, 512);

  // ---- GCN1 (512 -> 256) + BN + ReLU
  CVT(buf0, NN * 512);
  GEMM(abuf, wt_gcn1, nullptr, buf1, NN, 256, 512, 0, 0);
  FILL(buf2, 0.f, NN * 256);
  AGG(buf2, buf1, row_sl, col_sl, nullptr, dinv_sl, 1.f, 256, ESL);
  k_bn_stats<<<256, B256, 0, stream>>>(buf2, mu, rs, 256);
  k_bn_apply_relu<<<NN * 256 / 256, B256, 0, stream>>>(buf2, mu, rs, bn1_g, bn1_b, buf0, 256);

  // ---- SAGE: relu(mean@wl + bl + x@wr)         x = buf0
  FILL(buf1, 0.f, NN * 256);
  FILL(cnt, 0.f, NN);
  AGG(buf1, buf0, edge, edge + EE, nullptr, nullptr, 1.f, 256, EE);
  k_deg<<<cdiv(EE, 256), B256, 0, stream>>>(edge + EE, EE, cnt);
  k_sage_mean<<<NN * 256 / 256, B256, 0, stream>>>(buf1, cnt, 256);
  CVT(buf1, NN * 256);
  GEMM(abuf, wt_sl, sage_bl, buf2, NN, 256, 256, 0, 0);
  CVT(buf0, NN * 256);
  GEMM(abuf, wt_sr, nullptr, buf2, NN, 256, 256, 1, 1);     // += , fused ReLU

  // ---- Cheb (K=2): relu(x@w0 + tx1@w1 + b)     x = buf2
  FILL(deg, 0.f, NN);
  k_deg<<<cdiv(EE, 256), B256, 0, stream>>>(edge, EE, deg); // degree on row
  k_dinv<<<cdiv(NN, 256), B256, 0, stream>>>(deg, dinv_r, NN);
  FILL(buf1, 0.f, NN * 256);
  AGG(buf1, buf2, edge, edge + EE, nullptr, dinv_r, -1.f, 256, EE);  // tx1
  CVT(buf2, NN * 256);
  GEMM(abuf, wt_c0, cheb_b, buf0, NN, 256, 256, 0, 0);
  CVT(buf1, NN * 256);
  GEMM(abuf, wt_c1, nullptr, buf0, NN, 256, 256, 1, 1);     // += , fused ReLU

  // ---- GAT0: elu(softmax-agg(xw) + b)          x = buf0
  CVT(buf0, NN * 256);
  GEMM(abuf, wt_g0, nullptr, buf3, NN, 256, 256, 0, 0);
  k_gat_logits<<<NN / 8, B256, 0, stream>>>(buf3, gat0_as, gat0_ad, als, ald, 256);
  FILL(emax, -3.0e38f, NN);
  k_gat_e<<<cdiv(ESL, 256), B256, 0, stream>>>(als, ald, row_sl, col_sl, ev, emax, ESL);
  FILL(den, 0.f, NN);
  k_gat_exp<<<cdiv(ESL, 256), B256, 0, stream>>>(ev, col_sl, emax, den, ESL);
  k_gat_alpha<<<cdiv(ESL, 256), B256, 0, stream>>>(ev, col_sl, den, ESL);
  FILL(buf1, 0.f, NN * 256);
  AGG(buf1, buf3, row_sl, col_sl, ev, nullptr, 1.f, 256, ESL);
  k_bias_elu<<<NN * 256 / 256, B256, 0, stream>>>(buf1, gat0_b, 256);

  // ---- GAT1                                    x = buf1
  CVT(buf1, NN * 256);
  GEMM(abuf, wt_g1, nullptr, buf3, NN, 256, 256, 0, 0);
  k_gat_logits<<<NN / 8, B256, 0, stream>>>(buf3, gat1_as, gat1_ad, als, ald, 256);
  FILL(emax, -3.0e38f, NN);
  k_gat_e<<<cdiv(ESL, 256), B256, 0, stream>>>(als, ald, row_sl, col_sl, ev, emax, ESL);
  FILL(den, 0.f, NN);
  k_gat_exp<<<cdiv(ESL, 256), B256, 0, stream>>>(ev, col_sl, emax, den, ESL);
  k_gat_alpha<<<cdiv(ESL, 256), B256, 0, stream>>>(ev, col_sl, den, ESL);
  FILL(buf2, 0.f, NN * 256);
  AGG(buf2, buf3, row_sl, col_sl, ev, nullptr, 1.f, 256, ESL);
  k_bias_elu<<<NN * 256 / 256, B256, 0, stream>>>(buf2, gat1_b, 256);

  // ---- prediction head: [4096,256] x [256,41476] + pred_b  -> d_out
  CVT(buf2, NN * 256);
  GEMM(abuf, wt_pred, pred_b, (float*)d_out, NN, NCLS, 256, 0, 0);
}
